// prompt_generator_24326694764768
// MI455X (gfx1250) — matrix operand order repaired and checked by reading, not compiled, not run on hardware
//
#include <hip/hip_runtime.h>

// MI455X (gfx1250) fused 2-layer attention + final linear.
// bf16 WMMA (v_wmma_f32_16x16x32_bf16) with fp32 accumulate; ~21 GFLOP vs
// ~142MB HBM traffic -> compute-limited in fp32 VALU, so everything is routed
// through the matrix pipe. Weights are pre-swizzled into B-fragment order in
// LDS so every WMMA operand load is 2x ds_load_b128 (no scalar u16 gathers).
// All LDS pointers are addrspace(3)-qualified to force ds_* (round 1 showed
// flat_* leaking in when inference failed).

typedef __bf16 bf16_t;
typedef __attribute__((ext_vector_type(16))) __bf16 v16bf;
typedef __attribute__((ext_vector_type(8)))  float  v8f;

#define LDSQ __attribute__((address_space(3)))
#define GMQ  __attribute__((address_space(1)))

#define E_DIM 512
#define DK_DIM 32
#define SEQ 17
#define SPAD 32
#define NWAVES 4

// LDS layout (bytes):
//  [0,131072)           : Wq,Wk,Wv (B-frag order, 32 tiles*32 lanes*16) 32KB each,
//                         Dt (B-frag order, 32 tiles) 32KB
//  per wave (47232 B)   : X [32][512] bf16 row-major (32768)
//                         qb [32][32] row-major (2048)   -- A-frag source
//                         kf (B-frag order, 2 tiles) (2048)
//                         vf (B-frag order, 2 tiles) (2048)
//                         pb [32][32] row-major (2048)   -- A-frag source
//                         cb [32][32] row-major (2048)   -- A-frag source
//                         sc [32][33] f32 (4224)
#define LDS_TOTAL (131072 + NWAVES * 47232)  // = 320000 <= 320KB

__device__ __forceinline__ v8f wmma_bf16(v16bf a, v16bf b, v8f c) {
  return __builtin_amdgcn_wmma_f32_16x16x32_bf16(false, a, false, b, (short)0, c,
                                                 false, false);
}

// A fragment: 16x32 bf16 from row-major LDS. Per lane this is two contiguous
// 8-element (16B) runs -> compiler emits 2x ds_load_b128.
__device__ __forceinline__ v16bf load_a_frag(const LDSQ bf16_t* src, int ld, int r0,
                                             int c0) {
  const int lane = threadIdx.x & 31;
  const int hf = lane >> 4, m = lane & 15;
  const LDSQ bf16_t* row = src + (r0 + m) * ld + c0 + hf * 16;
  v16bf a;
#pragma unroll
  for (int idx = 0; idx < 8; ++idx) a[idx] = row[idx];
#pragma unroll
  for (int idx = 8; idx < 16; ++idx) a[idx] = row[idx];  // k = hf*16 + idx
  return a;
}

// B fragment from a buffer pre-swizzled into fragment order:
// element (lane, idx) at base[(tile*32 + lane)*16 + idx]  -> 2x ds_load_b128.
__device__ __forceinline__ v16bf load_b_packed(const LDSQ bf16_t* base, int tile) {
  const int lane = threadIdx.x & 31;
  const LDSQ bf16_t* p = base + (tile * 32 + lane) * 16;
  v16bf b;
#pragma unroll
  for (int idx = 0; idx < 16; ++idx) b[idx] = p[idx];
  return b;
}

// Stage one layer's weights into LDS in B-fragment order.
// QKV (B[k=e][n=d], tile = c*2+jt):  element(lane,idx) = w[(jt*16+n)*?]
//   e = c*32 + hf*16 + idx ; d = jt*16 + n ; value = w[d*512 + e]
// Dense (B[k=d][n=e], tile = et):    d = hf*16 + idx ; e = et*16 + n
__device__ void stage_weights(const GMQ float* wq, const GMQ float* wk,
                              const GMQ float* wv, const GMQ float* dw,
                              LDSQ bf16_t* Wq, LDSQ bf16_t* Wk, LDSQ bf16_t* Wv,
                              LDSQ bf16_t* Dt) {
  for (int i = threadIdx.x; i < E_DIM * DK_DIM; i += blockDim.x) {
    const int idx = i & 15;
    const int lane = (i >> 4) & 31;
    const int tile = i >> 9;  // 0..31
    const int hf = lane >> 4, n = lane & 15;
    {
      const int c = tile >> 1, jt = tile & 1;
      const int e = c * 32 + hf * 16 + idx;
      const int d = jt * 16 + n;
      Wq[i] = (bf16_t)wq[d * E_DIM + e];
      Wk[i] = (bf16_t)wk[d * E_DIM + e];
      Wv[i] = (bf16_t)wv[d * E_DIM + e];
    }
    {
      const int d = hf * 16 + idx;
      const int e = tile * 16 + n;
      Dt[i] = (bf16_t)dw[e * DK_DIM + d];
    }
  }
}

// One attention layer for one batch, executed by one wave. X ([32][512] bf16,
// row-major LDS, rows>=SEQ zero/garbage-but-masked) is overwritten with H.
__device__ void attn_layer(LDSQ bf16_t* X, LDSQ bf16_t* qb, LDSQ bf16_t* kf,
                           LDSQ bf16_t* vf, LDSQ bf16_t* pb, LDSQ bf16_t* cb,
                           LDSQ float* sc, const LDSQ bf16_t* Wq,
                           const LDSQ bf16_t* Wk, const LDSQ bf16_t* Wv,
                           const LDSQ bf16_t* Dt, const GMQ float* bq,
                           const GMQ float* bk, const GMQ float* bv,
                           const GMQ float* bd) {
  const int lane = threadIdx.x & 31;
  const int hf = lane >> 4, n = lane & 15;

  // ---- Q,K,V = X * W^T + b  (each 32x32, K=512 -> 16 WMMA per 16x16 tile) ----
#pragma unroll
  for (int mat = 0; mat < 3; ++mat) {
    const LDSQ bf16_t* W = (mat == 0) ? Wq : (mat == 1) ? Wk : Wv;
    const GMQ float* bias = (mat == 0) ? bq : (mat == 1) ? bk : bv;
    for (int it = 0; it < 2; ++it)
      for (int jt = 0; jt < 2; ++jt) {
        v8f acc = {};
        for (int c = 0; c < E_DIM / 32; ++c) {
          v16bf a = load_a_frag(X, E_DIM, it * 16, c * 32);
          v16bf b = load_b_packed(W, c * 2 + jt);
          acc = wmma_bf16(a, b, acc);
        }
        const float bv_ = bias[jt * 16 + n];
#pragma unroll
        for (int j = 0; j < 8; ++j) {
          const float val = acc[j] + bv_;
          const int jr = hf * 8 + j;        // row within 16-tile
          const int r = it * 16 + jr;       // global row (token)
          if (mat == 0) {
            qb[r * 32 + jt * 16 + n] = (bf16_t)val;  // row-major (A-frag source)
          } else if (mat == 1) {
            // kf B-frag order for scores: element (k=d, col=j):
            //   tile = j>>4 = it ; lane' = (j&15) + (d>>4)*16 = jr + jt*16 ; idx' = d&15 = n
            kf[((it * 32) + jr + jt * 16) * 16 + n] = (bf16_t)val;
          } else {
            // vf B-frag order for ctx: element (k=j, col=d):
            //   tile = d>>4 = jt ; lane' = n + (j>>4)*16 = n + it*16 ; idx' = j&15 = jr
            vf[((jt * 32) + n + it * 16) * 16 + jr] = (bf16_t)val;
          }
        }
      }
  }
  __syncthreads();

  // ---- scores = Q * K^T / sqrt(DK), mask padded cols ----
  const float scale = 0.17677669529663687f;  // 1/sqrt(32)
  for (int it = 0; it < 2; ++it)
    for (int jt = 0; jt < 2; ++jt) {
      v16bf a = load_a_frag(qb, 32, it * 16, 0);
      v16bf b = load_b_packed(kf, jt);
      v8f acc = {};
      acc = wmma_bf16(a, b, acc);
      const int col = jt * 16 + n;
#pragma unroll
      for (int j = 0; j < 8; ++j) {
        const int r = it * 16 + hf * 8 + j;
        sc[r * 33 + col] = (col < SEQ) ? acc[j] * scale : -1.0e30f;
      }
    }
  __syncthreads();

  // ---- softmax: lane L handles row L ----
  {
    float vrow[32];
    float mx = -1.0e30f;
#pragma unroll
    for (int j = 0; j < 32; ++j) {
      vrow[j] = sc[lane * 33 + j];
      mx = fmaxf(mx, vrow[j]);
    }
    float s = 0.f;
#pragma unroll
    for (int j = 0; j < 32; ++j) {
      vrow[j] = __expf(vrow[j] - mx);
      s += vrow[j];
    }
    const float inv = 1.0f / s;
#pragma unroll
    for (int j = 0; j < 32; ++j) pb[lane * 32 + j] = (bf16_t)(vrow[j] * inv);
  }
  __syncthreads();

  // ---- ctx = P * V  (32x32, K=32 -> 1 WMMA per tile) ----
  for (int it = 0; it < 2; ++it)
    for (int jt = 0; jt < 2; ++jt) {
      v16bf a = load_a_frag(pb, 32, it * 16, 0);
      v16bf b = load_b_packed(vf, jt);
      v8f acc = {};
      acc = wmma_bf16(a, b, acc);
#pragma unroll
      for (int j = 0; j < 8; ++j) {
        const int r = it * 16 + hf * 8 + j;
        cb[r * 32 + jt * 16 + n] = (bf16_t)acc[j];
      }
    }
  __syncthreads();

  // ---- H = ctx * dense^T + bd  -> overwrite X ----
  for (int it = 0; it < 2; ++it)
    for (int et = 0; et < E_DIM / 16; ++et) {
      v16bf a = load_a_frag(cb, 32, it * 16, 0);
      v16bf b = load_b_packed(Dt, et);
      v8f acc = {};
      acc = wmma_bf16(a, b, acc);
      const float bb = bd[et * 16 + n];
#pragma unroll
      for (int j = 0; j < 8; ++j) {
        const int r = it * 16 + hf * 8 + j;
        X[r * E_DIM + et * 16 + n] = (bf16_t)(acc[j] + bb);
      }
    }
  __syncthreads();
}

__global__ void __launch_bounds__(32 * NWAVES) attn_fused_kernel(
    const float* __restrict__ feature_map_, const float* __restrict__ cls_,
    const float* l1_wq_w, const float* l1_wq_b, const float* l1_wk_w,
    const float* l1_wk_b, const float* l1_wv_w, const float* l1_wv_b,
    const float* l1_dense_w, const float* l1_dense_b, const float* l2_wq_w,
    const float* l2_wq_b, const float* l2_wk_w, const float* l2_wk_b,
    const float* l2_wv_w, const float* l2_wv_b, const float* l2_dense_w,
    const float* l2_dense_b, bf16_t* __restrict__ cls_ws_) {
  extern __shared__ __align__(16) char smem_[];
  LDSQ char* smem = (LDSQ char*)smem_;
  LDSQ bf16_t* Wq = (LDSQ bf16_t*)smem;
  LDSQ bf16_t* Wk = Wq + E_DIM * DK_DIM;
  LDSQ bf16_t* Wv = Wk + E_DIM * DK_DIM;
  LDSQ bf16_t* Dt = Wv + E_DIM * DK_DIM;

  const int wave = threadIdx.x >> 5;
  const int lane = threadIdx.x & 31;
  LDSQ char* pw = smem + 131072 + wave * 47232;
  LDSQ bf16_t* X  = (LDSQ bf16_t*)pw;
  LDSQ bf16_t* qb = (LDSQ bf16_t*)(pw + 32768);
  LDSQ bf16_t* kf = (LDSQ bf16_t*)(pw + 34816);
  LDSQ bf16_t* vf = (LDSQ bf16_t*)(pw + 36864);
  LDSQ bf16_t* pb = (LDSQ bf16_t*)(pw + 38912);
  LDSQ bf16_t* cb = (LDSQ bf16_t*)(pw + 40960);
  LDSQ float*  sc = (LDSQ float*)(pw + 43008);

  const GMQ float* feature_map = (const GMQ float*)feature_map_;
  const GMQ float* cls = (const GMQ float*)cls_;
  GMQ bf16_t* cls_ws = (GMQ bf16_t*)cls_ws_;

  const int b = blockIdx.x * NWAVES + wave;

  // ---- stage X: row 0 = cls, rows 1..16 = feature_map[b], rows 17..31 = 0 ----
  for (int i = lane * 4; i < E_DIM; i += 128) {
#pragma unroll
    for (int u = 0; u < 4; ++u) X[i + u] = (bf16_t)cls[i + u];
  }
  const GMQ float* fm = feature_map + (size_t)b * (SEQ - 1) * E_DIM;
  for (int r = 1; r < SEQ; ++r)
    for (int i = lane * 4; i < E_DIM; i += 128) {
#pragma unroll
      for (int u = 0; u < 4; ++u)
        X[r * E_DIM + i + u] = (bf16_t)fm[(size_t)(r - 1) * E_DIM + i + u];
    }
  for (int r = SEQ; r < SPAD; ++r)
    for (int i = lane; i < E_DIM; i += 32) X[r * E_DIM + i] = (bf16_t)0.0f;

  // ---- layer 1 ----
  stage_weights((const GMQ float*)l1_wq_w, (const GMQ float*)l1_wk_w,
                (const GMQ float*)l1_wv_w, (const GMQ float*)l1_dense_w, Wq, Wk, Wv,
                Dt);
  __syncthreads();
  attn_layer(X, qb, kf, vf, pb, cb, sc, Wq, Wk, Wv, Dt, (const GMQ float*)l1_wq_b,
             (const GMQ float*)l1_wk_b, (const GMQ float*)l1_wv_b,
             (const GMQ float*)l1_dense_b);
  __syncthreads();

  // ---- layer 2 ----
  stage_weights((const GMQ float*)l2_wq_w, (const GMQ float*)l2_wk_w,
                (const GMQ float*)l2_wv_w, (const GMQ float*)l2_dense_w, Wq, Wk, Wv,
                Dt);
  __syncthreads();
  attn_layer(X, qb, kf, vf, pb, cb, sc, Wq, Wk, Wv, Dt, (const GMQ float*)l2_wq_b,
             (const GMQ float*)l2_wk_b, (const GMQ float*)l2_wv_b,
             (const GMQ float*)l2_dense_b);

  // ---- emit cls row (row 0 of H2) ----
  for (int i = lane; i < E_DIM; i += 32) cls_ws[(size_t)b * E_DIM + i] = X[i];
}

// out[b,e] = sum_f cls[b,f] * lin_w[e,f] + lin_b[e]
// One wave per 16x16 tile; K=512 -> 16 WMMAs. lin_w (1MB) stays L2-resident.
__global__ void __launch_bounds__(128) final_linear_kernel(
    const bf16_t* __restrict__ clsb_, const float* __restrict__ lin_w_,
    const float* __restrict__ lin_b_, float* __restrict__ out_) {
  const int wave = threadIdx.x >> 5;
  const int lane = threadIdx.x & 31;
  const int hf = lane >> 4, n = lane & 15;
  const int bt = blockIdx.x;             // 0..255 : batch tile of 16
  const int et = blockIdx.y * 4 + wave;  // 0..31  : 16 output features

  const GMQ bf16_t* clsb = (const GMQ bf16_t*)clsb_;
  const GMQ float* lin_w = (const GMQ float*)lin_w_;
  const GMQ float* lin_b = (const GMQ float*)lin_b_;
  GMQ float* out = (GMQ float*)out_;

  v8f acc = {};
  for (int kk = 0; kk < E_DIM; kk += 32) {
    // A-frag from global bf16 (row-major, one contiguous 32B run per lane)
    const GMQ bf16_t* arow =
        clsb + (size_t)(bt * 16 + n) * E_DIM + kk + hf * 16;
    v16bf a;
#pragma unroll
    for (int idx = 0; idx < 8; ++idx) a[idx] = arow[idx];
#pragma unroll
    for (int idx = 8; idx < 16; ++idx) a[idx] = arow[idx];
    // B[k=f][n=e] = lin_w[(et*16+n)*512 + kk + k], fp32 -> bf16 on the fly
    const GMQ float* src = lin_w + (size_t)(et * 16 + n) * E_DIM + kk + hf * 16;
    v16bf bfr;
#pragma unroll
    for (int idx = 0; idx < 16; ++idx) bfr[idx] = (bf16_t)src[idx];
    acc = wmma_bf16(a, bfr, acc);
  }
  const float bb = lin_b[et * 16 + n];
#pragma unroll
  for (int j = 0; j < 8; ++j) {
    const int row = bt * 16 + hf * 8 + j;
    out[(size_t)row * E_DIM + et * 16 + n] = acc[j] + bb;
  }
}

extern "C" void kernel_launch(void* const* d_in, const int* in_sizes, int n_in,
                              void* d_out, int out_size, void* d_ws, size_t ws_size,
                              hipStream_t stream) {
  const float* feature_map = (const float*)d_in[0];
  const float* cls         = (const float*)d_in[1];
  const float* lin_w       = (const float*)d_in[2];
  const float* lin_b       = (const float*)d_in[3];
  const float* l1_wq_w = (const float*)d_in[4];
  const float* l1_wq_b = (const float*)d_in[5];
  const float* l1_wk_w = (const float*)d_in[6];
  const float* l1_wk_b = (const float*)d_in[7];
  const float* l1_wv_w = (const float*)d_in[8];
  const float* l1_wv_b = (const float*)d_in[9];
  const float* l1_dense_w = (const float*)d_in[10];
  const float* l1_dense_b = (const float*)d_in[11];
  const float* l2_wq_w = (const float*)d_in[12];
  const float* l2_wq_b = (const float*)d_in[13];
  const float* l2_wk_w = (const float*)d_in[14];
  const float* l2_wk_b = (const float*)d_in[15];
  const float* l2_wv_w = (const float*)d_in[16];
  const float* l2_wv_b = (const float*)d_in[17];
  const float* l2_dense_w = (const float*)d_in[18];
  const float* l2_dense_b = (const float*)d_in[19];

  const int B = 4096;
  bf16_t* cls_ws = (bf16_t*)d_ws;  // B*512 bf16 = 4 MB

  attn_fused_kernel<<<dim3(B / NWAVES), dim3(32 * NWAVES), LDS_TOTAL, stream>>>(
      feature_map, cls, l1_wq_w, l1_wq_b, l1_wk_w, l1_wk_b, l1_wv_w, l1_wv_b,
      l1_dense_w, l1_dense_b, l2_wq_w, l2_wq_b, l2_wk_w, l2_wk_b, l2_wv_w, l2_wv_b,
      l2_dense_w, l2_dense_b, cls_ws);

  final_linear_kernel<<<dim3(B / 16, (E_DIM / 16) / 4), dim3(128), 0, stream>>>(
      cls_ws, lin_w, lin_b, (float*)d_out);
}
